// SymbolicTransformerModel_91147795955817
// MI455X (gfx1250) — matrix operand order repaired and checked by reading, not compiled, not run on hardware
//
#include <hip/hip_runtime.h>
#include <hip/hip_bf16.h>

// ---------------------------------------------------------------------------
// SymbolicTransformer on gfx1250 (MI455X).
// All big matmuls via v_wmma_f32_16x16x32_bf16 (fp32 accumulate).
// The V=32000-wide softmax paths are fused flash-style so the 393MB
// probability tensor is never materialized; each weight byte of
// vocab_w / wte is streamed from HBM exactly once per layer.
// Weight tiles are double-buffered through LDS: the next tile's HBM
// loads are in flight while the current tile runs through WMMA.
// ---------------------------------------------------------------------------

#define T_SEQ  256
#define NHEAD  12
#define HDIM   64
#define CDIM   768
#define VOCAB  32000
#define NSPLIT 20         // vocab splits: 240 workgroups of 16 waves

typedef __bf16 bf16_t;
typedef __attribute__((ext_vector_type(16))) __bf16 v16bf;
typedef __attribute__((ext_vector_type(8)))  float  v8f;

// native f32 -> bf16 (RNE); lowers to v_cvt_pk_bf16_f32 in pairs
__device__ __forceinline__ bf16_t f2bf(float f) { return (bf16_t)f; }

__device__ __forceinline__ v8f wmma_bf16(v16bf a, v16bf b, v8f c) {
  // D = A(16x32) * B(32x16) + C, wave32, fp32 accumulate
  return __builtin_amdgcn_wmma_f32_16x16x32_bf16(
      /*neg_a=*/false, a, /*neg_b=*/false, b,
      /*c_mod=*/(short)0, c, /*reuse_a=*/false, /*reuse_b=*/false);
}

// ---- fragment loaders (wave32; contiguous k order, same for A and B) -------------------------
// A operand (SRC0), 16x32 (MxK). Lane L: row = L&15;
//   elems 0..15 -> K = kb..kb+15, kb=(L>>4)*16
__device__ __forceinline__ v16bf frag_a_f32(const float* p, int ld, int lane) {
  int m  = lane & 15;
  int kb = (lane >> 4) * 16;
  const float* r = p + (long)m * ld;
  v16bf f;
#pragma unroll
  for (int j = 0; j < 8; ++j) { f[j] = f2bf(r[kb + j]); f[8 + j] = f2bf(r[kb + 8 + j]); }
  return f;
}
__device__ __forceinline__ v16bf frag_a_lds(const bf16_t* p, int ld, int lane) {
  int m  = lane & 15;
  int kb = (lane >> 4) * 16;
  const bf16_t* r = p + m * ld;
  v16bf f;
#pragma unroll
  for (int j = 0; j < 8; ++j) { f[j] = r[kb + j]; f[8 + j] = r[kb + 8 + j]; }
  return f;
}
// B operand (SRC1), 32x16 (KxN). Lane L: col n = L&15, K = base..base+15,
// base = (L>>4)*16.  Source stored as Bt[n][k] row-major (weights are [N,K]).
__device__ __forceinline__ v16bf frag_bt_f32(const float* p, int ld, int lane) {
  int n = lane & 15;
  int kb = (lane >> 4) * 16;
  const float* r = p + (long)n * ld + kb;
  v16bf f;
#pragma unroll
  for (int j = 0; j < 16; ++j) f[j] = f2bf(r[j]);
  return f;
}
__device__ __forceinline__ v16bf frag_bt_lds(const bf16_t* p, int ld, int lane) {
  int n = lane & 15;
  int kb = (lane >> 4) * 16;
  const bf16_t* r = p + n * ld + kb;
  v16bf f;
#pragma unroll
  for (int j = 0; j < 16; ++j) f[j] = r[j];
  return f;
}
// B operand from B[k][n] row-major (column reads): used for V-tiles / E-tiles.
__device__ __forceinline__ v16bf frag_b_f32(const float* p, int ld, int lane) {
  int n = lane & 15;
  int kb = (lane >> 4) * 16;
  v16bf f;
#pragma unroll
  for (int j = 0; j < 16; ++j) f[j] = f2bf(p[(long)(kb + j) * ld + n]);
  return f;
}
__device__ __forceinline__ v16bf frag_b_lds(const bf16_t* p, int ld, int lane) {
  int n = lane & 15;
  int kb = (lane >> 4) * 16;
  v16bf f;
#pragma unroll
  for (int j = 0; j < 16; ++j) f[j] = p[(kb + j) * ld + n];
  return f;
}

// half-group (16 lanes sharing a C-tile row-set) reductions, wave32
__device__ __forceinline__ float hgmax(float v) {
  v = fmaxf(v, __shfl_xor(v, 1, 32));
  v = fmaxf(v, __shfl_xor(v, 2, 32));
  v = fmaxf(v, __shfl_xor(v, 4, 32));
  v = fmaxf(v, __shfl_xor(v, 8, 32));
  return v;
}
__device__ __forceinline__ float hgsum(float v) {
  v += __shfl_xor(v, 1, 32);
  v += __shfl_xor(v, 2, 32);
  v += __shfl_xor(v, 4, 32);
  v += __shfl_xor(v, 8, 32);
  return v;
}

// ---------------------------------------------------------------------------
// Embedding gather: x[t, c] = wte[idx[t], c]
__global__ void k_embed(const int* __restrict__ idx, const float* __restrict__ wte,
                        float* __restrict__ x) {
  int i = blockIdx.x * blockDim.x + threadIdx.x;
  if (i >= T_SEQ * CDIM) return;
  int t = i / CDIM, c = i % CDIM;
  x[i] = wte[(long)idx[t] * CDIM + c];
}

// Per-head LayerNorm over D=64.  One thread per (t, h).
__global__ void k_headln(const float* __restrict__ x, const float* __restrict__ w,
                         const float* __restrict__ b, float* __restrict__ y) {
  int i = blockIdx.x * blockDim.x + threadIdx.x;
  if (i >= T_SEQ * NHEAD) return;
  int t = i / NHEAD, h = i % NHEAD;
  const float* p = x + (long)t * CDIM + h * HDIM;
  float mu = 0.f;
#pragma unroll 8
  for (int j = 0; j < HDIM; ++j) mu += p[j];
  mu *= (1.0f / HDIM);
  float var = 0.f;
#pragma unroll 8
  for (int j = 0; j < HDIM; ++j) { float d = p[j] - mu; var += d * d; }
  var *= (1.0f / HDIM);
  float inv = rsqrtf(var + 1e-5f);
  float* o = y + (long)t * CDIM + h * HDIM;
  float wh = w[h], bh = b[h];
#pragma unroll 8
  for (int j = 0; j < HDIM; ++j) o[j] = (p[j] - mu) * inv * wh + bh;
}

// ---------------------------------------------------------------------------
// Generic WMMA GEMM:  Out[M=256, n0:n0+64] = A[256,K] * W[N,K]^T (+bias)(+res)
// 8 waves; wave w owns rows [32w, 32w+32) as two 16-row tiles.
// W-tile (64n x 32k) double-buffered fp32->bf16 through LDS.
__global__ __launch_bounds__(256) void k_gemm(
    const float* __restrict__ A, int lda, long sA,
    const float* __restrict__ W, int ldw, long sW,
    const float* __restrict__ bias, long sBias,
    const float* __restrict__ res, int ldr, long sRes,
    float* __restrict__ Out, int ldo, long sO, int K) {
  __shared__ bf16_t ldsW[2][64 * 40];        // [n][k], k padded 32->40
  int tid = threadIdx.x, lane = tid & 31, wave = tid >> 5;
  int bz = blockIdx.z;
  int n0 = blockIdx.x * 64;
  const float* Ab = A + (long)bz * sA;
  const float* Wb = W + (long)bz * sW + (long)n0 * ldw;
  v8f acc[2][4];
#pragma unroll
  for (int s = 0; s < 2; ++s)
#pragma unroll
    for (int n = 0; n < 4; ++n)
#pragma unroll
      for (int r = 0; r < 8; ++r) acc[s][n][r] = 0.f;

  int sn = tid >> 3;                 // 0..31 (staging row)
  int sk = (tid & 7) * 4;            // 0..28 (staging k, float4)
  const float* Wr0 = Wb + (long)sn * ldw + sk;
  const float* Wr1 = Wb + (long)(sn + 32) * ldw + sk;

  // prologue: stage tile 0 into buffer 0
  float4 r0 = *(const float4*)(Wr0);
  float4 r1 = *(const float4*)(Wr1);
  {
    bf16_t* d0 = &ldsW[0][sn * 40 + sk];
    bf16_t* d1 = &ldsW[0][(sn + 32) * 40 + sk];
    d0[0] = f2bf(r0.x); d0[1] = f2bf(r0.y); d0[2] = f2bf(r0.z); d0[3] = f2bf(r0.w);
    d1[0] = f2bf(r1.x); d1[1] = f2bf(r1.y); d1[2] = f2bf(r1.z); d1[3] = f2bf(r1.w);
  }
  int buf = 0;
  for (int k0 = 0; k0 < K; k0 += 32) {
    __syncthreads();
    bool more = (k0 + 32) < K;
    if (more) {                      // next tile's HBM loads fly over compute
      r0 = *(const float4*)(Wr0 + k0 + 32);
      r1 = *(const float4*)(Wr1 + k0 + 32);
    }
#pragma unroll
    for (int sub = 0; sub < 2; ++sub) {
      int m0 = wave * 32 + sub * 16;
      v16bf a = frag_a_f32(Ab + (long)m0 * lda + k0, lda, lane);
#pragma unroll
      for (int nt = 0; nt < 4; ++nt) {
        v16bf bf = frag_bt_lds(&ldsW[buf][nt * 16 * 40], 40, lane);
        acc[sub][nt] = wmma_bf16(a, bf, acc[sub][nt]);
      }
    }
    if (more) {
      bf16_t* d0 = &ldsW[buf ^ 1][sn * 40 + sk];
      bf16_t* d1 = &ldsW[buf ^ 1][(sn + 32) * 40 + sk];
      d0[0] = f2bf(r0.x); d0[1] = f2bf(r0.y); d0[2] = f2bf(r0.z); d0[3] = f2bf(r0.w);
      d1[0] = f2bf(r1.x); d1[1] = f2bf(r1.y); d1[2] = f2bf(r1.z); d1[3] = f2bf(r1.w);
    }
    buf ^= 1;
  }
  int hi = lane >> 4, cl = lane & 15;
#pragma unroll
  for (int sub = 0; sub < 2; ++sub) {
    int m0 = wave * 32 + sub * 16;
#pragma unroll
    for (int nt = 0; nt < 4; ++nt) {
      int col = n0 + nt * 16 + cl;
      float bv = bias ? bias[bz * sBias + col] : 0.f;
#pragma unroll
      for (int r = 0; r < 8; ++r) {
        int row = m0 + r + 8 * hi;
        float v = acc[sub][nt][r] + bv;
        if (res) v += res[(long)bz * sRes + (long)row * ldr + col];
        Out[(long)bz * sO + (long)row * ldo + col] = v;
      }
    }
  }
}

// ---------------------------------------------------------------------------
// Flash attention: 1 wave per (head, 16-query tile). ALiBi + causal mask,
// online softmax, S and P*V through WMMA.  qkv layout: [t][3C], q|k|v blocks.
__global__ __launch_bounds__(32) void k_attn(const float* __restrict__ qkv,
                                             float* __restrict__ y) {
  __shared__ bf16_t ldsP[16 * 40];
  int lane = threadIdx.x;
  int h = blockIdx.y, q0 = blockIdx.x * 16;
  float slope = (h < 8) ? exp2f(-(float)(h + 1)) : exp2f(-0.5f * (float)(h - 7));
  const float* Q = qkv + (long)q0 * (3 * CDIM) + h * HDIM;
  v16bf a0 = frag_a_f32(Q, 3 * CDIM, lane);
  v16bf a1 = frag_a_f32(Q + 32, 3 * CDIM, lane);
  v8f acc[4];
  float m_run[8], s_run[8];
#pragma unroll
  for (int d = 0; d < 4; ++d)
#pragma unroll
    for (int r = 0; r < 8; ++r) acc[d][r] = 0.f;
#pragma unroll
  for (int r = 0; r < 8; ++r) { m_run[r] = -3.0e38f; s_run[r] = 0.f; }
  int hi = lane >> 4, cl = lane & 15;

  for (int kb = 0; kb <= q0 + 15; kb += 32) {
    v8f S[2];
#pragma unroll
    for (int nt = 0; nt < 2; ++nt) {
      const float* Kp = qkv + (long)(kb + nt * 16) * (3 * CDIM) + CDIM + h * HDIM;
      v16bf b0 = frag_bt_f32(Kp, 3 * CDIM, lane);
      v16bf b1 = frag_bt_f32(Kp + 32, 3 * CDIM, lane);
      v8f s;
#pragma unroll
      for (int r = 0; r < 8; ++r) s[r] = 0.f;
      s = wmma_bf16(a0, b0, s);
      s = wmma_bf16(a1, b1, s);
      S[nt] = s;
    }
    // scale + ALiBi + causal mask (C-layout: row = r+8*hi, col = cl)
#pragma unroll
    for (int nt = 0; nt < 2; ++nt)
#pragma unroll
      for (int r = 0; r < 8; ++r) {
        int ki = kb + nt * 16 + cl;
        int qi = q0 + r + 8 * hi;
        float v = S[nt][r] * 0.125f + slope * (float)(ki - qi);
        S[nt][r] = (ki <= qi) ? v : -3.0e38f;
      }
    float mnew[8], scale[8];
#pragma unroll
    for (int r = 0; r < 8; ++r) {
      float v = hgmax(fmaxf(S[0][r], S[1][r]));
      mnew[r]  = fmaxf(m_run[r], v);
      scale[r] = __expf(m_run[r] - mnew[r]);
      m_run[r] = mnew[r];
    }
#pragma unroll
    for (int nt = 0; nt < 2; ++nt)
#pragma unroll
      for (int r = 0; r < 8; ++r) S[nt][r] = __expf(S[nt][r] - mnew[r]);
#pragma unroll
    for (int r = 0; r < 8; ++r)
      s_run[r] = s_run[r] * scale[r] + hgsum(S[0][r] + S[1][r]);
#pragma unroll
    for (int d = 0; d < 4; ++d)
#pragma unroll
      for (int r = 0; r < 8; ++r) acc[d][r] *= scale[r];
    // P: C-layout -> A-layout via LDS round trip
#pragma unroll
    for (int r = 0; r < 8; ++r) {
      int row = r + 8 * hi;
      ldsP[row * 40 + cl]      = f2bf(S[0][r]);
      ldsP[row * 40 + 16 + cl] = f2bf(S[1][r]);
    }
    __syncthreads();
    v16bf pf = frag_a_lds(ldsP, 40, lane);
#pragma unroll
    for (int d = 0; d < 4; ++d) {
      const float* Vp = qkv + (long)kb * (3 * CDIM) + 2 * CDIM + h * HDIM + d * 16;
      v16bf bv = frag_b_f32(Vp, 3 * CDIM, lane);
      acc[d] = wmma_bf16(pf, bv, acc[d]);
    }
    __syncthreads();
  }
#pragma unroll
  for (int d = 0; d < 4; ++d)
#pragma unroll
    for (int r = 0; r < 8; ++r) {
      int row = q0 + r + 8 * hi;
      y[(long)row * CDIM + h * HDIM + d * 16 + cl] = acc[d][r] / s_run[r];
    }
}

// ---------------------------------------------------------------------------
// Fused vocab flash: per (head, V-split) workgroup of 16 waves = all 256 q.
// logits = h1 * vocab_w^T  ->  online softmax/temp  ->  acc += P * E
// Weight bytes stream through double-buffered LDS exactly once per pass;
// the next 32-row tile's HBM loads are issued before the current tile's
// WMMA/softmax work so the stream never stalls on memory.
__global__ __launch_bounds__(512) void k_vocab(
    const float* __restrict__ h1, const float* __restrict__ vw,
    const float* __restrict__ wte, const float* __restrict__ temps,
    float* __restrict__ part, int vPerSplit) {
  __shared__ bf16_t ldsW[2][32 * 72];   // vocab_w tile  [v][k=64], pad 72
  __shared__ bf16_t ldsE[2][32 * 72];   // embed  tile   [v][d=64], pad 72
  __shared__ bf16_t ldsP[16 * 16 * 40];
  int tid = threadIdx.x, lane = tid & 31, wave = tid >> 5;
  int h = blockIdx.z, split = blockIdx.x;
  int q0 = wave * 16;
  float invt = 1.0f / fmaxf(temps[h], 0.1f);
  const float* Aq = h1 + (long)q0 * CDIM + h * HDIM;
  v16bf a0 = frag_a_f32(Aq, CDIM, lane);
  v16bf a1 = frag_a_f32(Aq + 32, CDIM, lane);
  const float* Wh = vw + (long)h * VOCAB * HDIM;
  bf16_t* myP = ldsP + wave * 16 * 40;
  v8f acc[4];
  float m_run[8], s_run[8];
#pragma unroll
  for (int d = 0; d < 4; ++d)
#pragma unroll
    for (int r = 0; r < 8; ++r) acc[d][r] = 0.f;
#pragma unroll
  for (int r = 0; r < 8; ++r) { m_run[r] = -3.0e38f; s_run[r] = 0.f; }
  int hi = lane >> 4, cl = lane & 15;
  int v0 = split * vPerSplit;
  int svi = tid >> 4;                // 0..31 (staging row)
  int sk  = (tid & 15) * 4;          // 0..60 (staging k, float4)

  // prologue: stage tile v0 into buffer 0
  float4 w4 = *(const float4*)(&Wh[(long)(v0 + svi) * HDIM + sk]);
  float4 e4 = *(const float4*)(&wte[(long)(v0 + svi) * CDIM + h * HDIM + sk]);
  {
    bf16_t* dw = &ldsW[0][svi * 72 + sk];
    bf16_t* de = &ldsE[0][svi * 72 + sk];
    dw[0] = f2bf(w4.x); dw[1] = f2bf(w4.y); dw[2] = f2bf(w4.z); dw[3] = f2bf(w4.w);
    de[0] = f2bf(e4.x); de[1] = f2bf(e4.y); de[2] = f2bf(e4.z); de[3] = f2bf(e4.w);
  }
  int buf = 0;
  for (int vb = v0; vb < v0 + vPerSplit; vb += 32) {
    __syncthreads();
    bool more = (vb + 32) < (v0 + vPerSplit);
    if (more) {                      // next tile's HBM loads fly over compute
      w4 = *(const float4*)(&Wh[(long)(vb + 32 + svi) * HDIM + sk]);
      e4 = *(const float4*)(&wte[(long)(vb + 32 + svi) * CDIM + h * HDIM + sk]);
      __builtin_prefetch(&Wh[(long)(vb + 64 + svi) * HDIM], 0, 1);
      __builtin_prefetch(&wte[(long)(vb + 64 + svi) * CDIM + h * HDIM], 0, 1);
    }
    v8f S[2];
#pragma unroll
    for (int nt = 0; nt < 2; ++nt) {
      v16bf b0 = frag_bt_lds(&ldsW[buf][nt * 16 * 72], 72, lane);
      v16bf b1 = frag_bt_lds(&ldsW[buf][nt * 16 * 72 + 32], 72, lane);
      v8f s;
#pragma unroll
      for (int r = 0; r < 8; ++r) s[r] = 0.f;
      s = wmma_bf16(a0, b0, s);
      s = wmma_bf16(a1, b1, s);
#pragma unroll
      for (int r = 0; r < 8; ++r) s[r] *= invt;
      S[nt] = s;
    }
    float mnew[8], scale[8];
#pragma unroll
    for (int r = 0; r < 8; ++r) {
      float v = hgmax(fmaxf(S[0][r], S[1][r]));
      mnew[r]  = fmaxf(m_run[r], v);
      scale[r] = __expf(m_run[r] - mnew[r]);
      m_run[r] = mnew[r];
    }
#pragma unroll
    for (int nt = 0; nt < 2; ++nt)
#pragma unroll
      for (int r = 0; r < 8; ++r) S[nt][r] = __expf(S[nt][r] - mnew[r]);
#pragma unroll
    for (int r = 0; r < 8; ++r)
      s_run[r] = s_run[r] * scale[r] + hgsum(S[0][r] + S[1][r]);
#pragma unroll
    for (int d = 0; d < 4; ++d)
#pragma unroll
      for (int r = 0; r < 8; ++r) acc[d][r] *= scale[r];
#pragma unroll
    for (int r = 0; r < 8; ++r) {          // per-wave LDS region: no barrier
      int row = r + 8 * hi;
      myP[row * 40 + cl]      = f2bf(S[0][r]);
      myP[row * 40 + 16 + cl] = f2bf(S[1][r]);
    }
    v16bf pf = frag_a_lds(myP, 40, lane);
#pragma unroll
    for (int d = 0; d < 4; ++d) {
      v16bf be = frag_b_lds(&ldsE[buf][d * 16], 72, lane);
      acc[d] = wmma_bf16(pf, be, acc[d]);
    }
    if (more) {                      // convert + park next tile in other buffer
      bf16_t* dw = &ldsW[buf ^ 1][svi * 72 + sk];
      bf16_t* de = &ldsE[buf ^ 1][svi * 72 + sk];
      dw[0] = f2bf(w4.x); dw[1] = f2bf(w4.y); dw[2] = f2bf(w4.z); dw[3] = f2bf(w4.w);
      de[0] = f2bf(e4.x); de[1] = f2bf(e4.y); de[2] = f2bf(e4.z); de[3] = f2bf(e4.w);
    }
    buf ^= 1;
  }
  long base = ((long)(h * NSPLIT + split) * T_SEQ + q0) * 66;
#pragma unroll
  for (int d = 0; d < 4; ++d)
#pragma unroll
    for (int r = 0; r < 8; ++r) {
      int row = r + 8 * hi;
      part[base + (long)row * 66 + 2 + d * 16 + cl] = acc[d][r];
    }
  if (cl == 0) {
#pragma unroll
    for (int r = 0; r < 8; ++r) {
      int row = r + 8 * hi;
      part[base + (long)row * 66 + 0] = m_run[r];
      part[base + (long)row * 66 + 1] = s_run[r];
    }
  }
}

// Combine V-split partials and add residual into x.
__global__ void k_vreduce(const float* __restrict__ part, float* __restrict__ x) {
  int q = blockIdx.x, h = blockIdx.y, d = threadIdx.x;
  float M = -3.0e38f;
  for (int s = 0; s < NSPLIT; ++s)
    M = fmaxf(M, part[((long)(h * NSPLIT + s) * T_SEQ + q) * 66 + 0]);
  float stot = 0.f, a = 0.f;
  for (int s = 0; s < NSPLIT; ++s) {
    long b = ((long)(h * NSPLIT + s) * T_SEQ + q) * 66;
    float w = __expf(part[b + 0] - M);
    stot += part[b + 1] * w;
    a    += part[b + 2 + d] * w;
  }
  x[(long)q * CDIM + h * HDIM + d] += a / stot;
}

// ---------------------------------------------------------------------------
extern "C" void kernel_launch(void* const* d_in, const int* in_sizes, int n_in,
                              void* d_out, int out_size, void* d_ws, size_t ws_size,
                              hipStream_t stream) {
  const int*   idx    = (const int*)  d_in[0];
  const float* wte    = (const float*)d_in[1];
  const float* qkv_w  = (const float*)d_in[2];
  const float* qkv_b  = (const float*)d_in[3];
  const float* proj_w = (const float*)d_in[4];
  const float* proj_b = (const float*)d_in[5];
  const float* ln1_w  = (const float*)d_in[6];
  const float* ln1_b  = (const float*)d_in[7];
  const float* ln2_w  = (const float*)d_in[8];
  const float* ln2_b  = (const float*)d_in[9];
  const float* ffn_w  = (const float*)d_in[10];
  const float* ffn_b  = (const float*)d_in[11];
  const float* voc_w  = (const float*)d_in[12];
  const float* temps  = (const float*)d_in[13];
  const float* lnf_w  = (const float*)d_in[14];
  const float* lnf_b  = (const float*)d_in[15];
  float* out = (float*)d_out;

  float* ws   = (float*)d_ws;
  float* x    = ws;                              // 256*768
  float* y    = x    + T_SEQ * CDIM;             // 256*768
  float* qkv  = y    + T_SEQ * CDIM;             // 256*2304
  float* atty = qkv  + T_SEQ * 3 * CDIM;         // 256*768
  float* h1   = atty + T_SEQ * CDIM;             // 256*768
  float* part = h1   + T_SEQ * CDIM;             // 12*20*256*66 (~16.2MB)

  k_embed<<<(T_SEQ * CDIM + 255) / 256, 256, 0, stream>>>(idx, wte, x);

  for (int l = 0; l < 2; ++l) {
    k_headln<<<24, 128, 0, stream>>>(x, ln1_w + l * NHEAD, ln1_b + l * NHEAD, y);
    // qkv = ln1(x) @ qkv_w^T + qkv_b
    k_gemm<<<dim3(3 * CDIM / 64, 1, 1), 256, 0, stream>>>(
        y, CDIM, 0, qkv_w + (long)l * 3 * CDIM * CDIM, CDIM, 0,
        qkv_b + (long)l * 3 * CDIM, 0, nullptr, 0, 0, qkv, 3 * CDIM, 0, CDIM);
    k_attn<<<dim3(T_SEQ / 16, NHEAD), 32, 0, stream>>>(qkv, atty);
    // x += atty @ proj_w^T + proj_b
    k_gemm<<<dim3(CDIM / 64, 1, 1), 256, 0, stream>>>(
        atty, CDIM, 0, proj_w + (long)l * CDIM * CDIM, CDIM, 0,
        proj_b + (long)l * CDIM, 0, x, CDIM, 0, x, CDIM, 0, CDIM);
    k_headln<<<24, 128, 0, stream>>>(x, ln2_w + l * NHEAD, ln2_b + l * NHEAD, y);
    // h1[t,h,:] = ln2(x)[t,h,:] @ ffn_w[l,h]^T + ffn_b   (batched over heads)
    k_gemm<<<dim3(1, 1, NHEAD), 256, 0, stream>>>(
        y, CDIM, HDIM, ffn_w + (long)l * NHEAD * HDIM * HDIM, HDIM, HDIM * HDIM,
        ffn_b + (long)l * NHEAD * HDIM, HDIM, nullptr, 0, 0, h1, CDIM, HDIM, HDIM);
    k_vocab<<<dim3(NSPLIT, 1, NHEAD), 512, 0, stream>>>(
        h1, voc_w + (long)l * NHEAD * VOCAB * HDIM, wte, temps + l * NHEAD,
        part, VOCAB / NSPLIT);
    k_vreduce<<<dim3(T_SEQ, NHEAD), HDIM, 0, stream>>>(part, x);
  }
  k_headln<<<24, 128, 0, stream>>>(x, lnf_w, lnf_b, y);
  // logits = lnf(x) @ wte^T
  k_gemm<<<dim3(VOCAB / 64, 1, 1), 256, 0, stream>>>(
      y, CDIM, 0, wte, CDIM, 0, nullptr, 0, nullptr, 0, 0, out, VOCAB, 0, CDIM);
}
